// _NonLocalNd_bn_75591424410039
// MI455X (gfx1250) — matrix operand order repaired and checked by reading, not compiled, not run on hardware
//
#include <hip/hip_runtime.h>
#include <hip/hip_bf16.h>
#include <math.h>

// ---------------------------------------------------------------------------
// NonLocal block (frozen-BN) for MI455X / gfx1250.
// Heavy math on v_wmma_f32_16x16x32_bf16; LDS staging via async-to-LDS copies
// (ASYNCcnt) where the global layout is K-contiguous, double-buffered.
// ---------------------------------------------------------------------------

typedef __attribute__((ext_vector_type(8)))  float  v8f;
typedef __attribute__((ext_vector_type(16))) __bf16 v16bf;
typedef __attribute__((ext_vector_type(4)))  int    v4i;

union Frag16 { v16bf v; unsigned int u[8]; };

#define ASG __attribute__((address_space(1)))
#define ASL __attribute__((address_space(3)))

#if defined(__AMDGCN__) && __has_builtin(__builtin_amdgcn_global_load_async_to_lds_b128) && \
    __has_builtin(__builtin_amdgcn_s_wait_asynccnt)
#define ASYNC_OK 1
#else
#define ASYNC_OK 0
#endif

__device__ __forceinline__ void async_copy_b128(const void* g, void* l) {
#if ASYNC_OK
  __builtin_amdgcn_global_load_async_to_lds_b128((ASG v4i*)g, (ASL v4i*)l, 0, 0);
#else
  *reinterpret_cast<uint4*>(l) = *reinterpret_cast<const uint4*>(g);
#endif
}

__device__ __forceinline__ void tile_fence() {
#if ASYNC_OK
  __builtin_amdgcn_s_wait_asynccnt(0);
#endif
  __syncthreads();
}

__device__ __forceinline__ unsigned short f2bf(float f) {
  unsigned int u = __float_as_uint(f);
  u += 0x7FFFu + ((u >> 16) & 1u);            // round-to-nearest-even
  return (unsigned short)(u >> 16);
}

// ---- fp32 -> bf16 bulk convert (vectorized x4) ----------------------------
__global__ __launch_bounds__(256) void cvt_f32_bf16_vec4(
    const float* __restrict__ src, unsigned short* __restrict__ dst, int n4) {
  int i = blockIdx.x * blockDim.x + threadIdx.x;
  if (i >= n4) return;
  float4 v = reinterpret_cast<const float4*>(src)[i];
  ushort4 o;
  o.x = f2bf(v.x); o.y = f2bf(v.y); o.z = f2bf(v.z); o.w = f2bf(v.w);
  reinterpret_cast<ushort4*>(dst)[i] = o;
}

// ---- 2x2 maxpool (64x64 -> 32x32), output bf16 ----------------------------
__global__ __launch_bounds__(256) void pool2x2_bf16(
    const float* __restrict__ x, unsigned short* __restrict__ out, int total) {
  int t = blockIdx.x * blockDim.x + threadIdx.x;
  if (t >= total) return;
  int j  = t & 31;
  int i  = (t >> 5) & 31;
  int bc = t >> 10;                            // fused (b*C + c)
  const float* p = x + ((unsigned)(bc * 64 + 2 * i) * 64u + (unsigned)(2 * j));
  float2 a = *reinterpret_cast<const float2*>(p);
  float2 b = *reinterpret_cast<const float2*>(p + 64);
  out[t] = f2bf(fmaxf(fmaxf(a.x, a.y), fmaxf(b.x, b.y)));
}

// ---- fold frozen BN into per-row scale/bias -------------------------------
__global__ void bn_fold(const float* __restrict__ g, const float* __restrict__ beta,
                        const float* __restrict__ mean, const float* __restrict__ var,
                        const float* __restrict__ convb,
                        float* __restrict__ s, float* __restrict__ t, int n) {
  int i = blockIdx.x * blockDim.x + threadIdx.x;
  if (i >= n) return;
  float sc = g[i] * rsqrtf(var[i] + 1e-5f);
  s[i] = sc;
  t[i] = (convb[i] - mean[i]) * sc + beta[i];
}

__global__ void fill_const(float* __restrict__ p, float v, int n) {
  int i = blockIdx.x * blockDim.x + threadIdx.x;
  if (i < n) p[i] = v;
}

// ---------------------------------------------------------------------------
// Batched bf16 WMMA GEMM.  C[M,N] = op(A) * op(B)
//   TA=0: A is [M,K] row-major   TA=1: A is [K,M] row-major (A^T used)
//   TB=0: B is [K,N] row-major   TB=1: B is [N,K] row-major (B^T used)
// Block tile 64x128, BK=32, 8 waves (2x4), each wave 32x32 via 2x2 WMMAs.
// LDS tiles K-major (stride 40), double-buffered; async-to-LDS when the
// global layout is K-contiguous (TA=0 for A, TB=1 for B).
// EPI: 0 -> bf16 out, per-row scale/bias (must be valid pointers)
//      1 -> fp32 out * constScale
//      3 -> fp32 out = residual + gamma * acc
// ---------------------------------------------------------------------------
template <bool TA, bool TB, int EPI>
__global__ __launch_bounds__(256, 2) void gemm_bf16_wmma(
    const unsigned short* __restrict__ A, const unsigned short* __restrict__ B,
    unsigned short* __restrict__ Cb, float* __restrict__ Cf,
    int Mdim, int Ndim, int Kdim,
    unsigned strideA, unsigned strideB, unsigned strideC,
    const float* __restrict__ rowScale, const float* __restrict__ rowBias,
    float constScale,
    const float* __restrict__ residual, unsigned strideR,
    const float* __restrict__ gammaPtr) {
  constexpr int LDT = 40;                      // padded K-major LDS row stride
  __shared__ unsigned short As[2][64 * LDT];   // [m][k]
  __shared__ unsigned short Bs[2][128 * LDT];  // [n][k]
  __shared__ float sSc[64], sBi[64];

  const int tid  = threadIdx.x;
  const int lane = tid & 31;
  const int wave = tid >> 5;
  const int wm   = wave >> 2;                  // 0..1
  const int wn   = wave & 3;                   // 0..3
  const int half = lane >> 4;                  // 0..1
  const int l16  = lane & 15;

  const int bm0 = blockIdx.y * 64;
  const int bn0 = blockIdx.x * 128;
  const int b   = blockIdx.z;
  A += (unsigned)b * strideA;
  B += (unsigned)b * strideB;

  if (EPI == 0) {
    if (tid < 64) {
      sSc[tid] = rowScale[bm0 + tid];
      sBi[tid] = rowBias[bm0 + tid];
    }
  }

  // ---- tile staging -------------------------------------------------------
  auto stageA = [&](int buf, int k0) {
    if (!TA) {                                 // A[M,K]: K-contiguous -> async
      int i = tid >> 2, c = (tid & 3) << 3;
      const unsigned short* src = A + (unsigned)(bm0 + i) * (unsigned)Kdim + (unsigned)(k0 + c);
      async_copy_b128(src, &As[buf][i * LDT + c]);
    } else {                                   // A[K,M]: transpose scatter
      int kk = tid & 31, i0 = (tid >> 5) << 3;
      const unsigned short* src = A + (unsigned)(k0 + kk) * (unsigned)Mdim + (unsigned)(bm0 + i0);
      uint4 v = *reinterpret_cast<const uint4*>(src);
      const unsigned short* pv = reinterpret_cast<const unsigned short*>(&v);
#pragma unroll
      for (int e = 0; e < 8; ++e) As[buf][(i0 + e) * LDT + kk] = pv[e];
      __builtin_prefetch(src + 32u * (unsigned)Mdim, 0, 1);
    }
  };
  auto stageB = [&](int buf, int k0) {
    if (!TB) {                                 // B[K,N]: transpose scatter
      int kk = tid >> 3, j0 = (tid & 7) << 4;
      const unsigned short* src = B + (unsigned)(k0 + kk) * (unsigned)Ndim + (unsigned)(bn0 + j0);
      uint4 v0 = reinterpret_cast<const uint4*>(src)[0];
      uint4 v1 = reinterpret_cast<const uint4*>(src)[1];
      const unsigned short* pv = reinterpret_cast<const unsigned short*>(&v0);
#pragma unroll
      for (int e = 0; e < 8; ++e) Bs[buf][(j0 + e) * LDT + kk] = pv[e];
      pv = reinterpret_cast<const unsigned short*>(&v1);
#pragma unroll
      for (int e = 0; e < 8; ++e) Bs[buf][(j0 + 8 + e) * LDT + kk] = pv[e];
      __builtin_prefetch(src + 32u * (unsigned)Ndim, 0, 1);
    } else {                                   // B[N,K]: K-contiguous -> async
      int j = tid >> 1, kk0 = (tid & 1) << 4;
      const unsigned short* src = B + (unsigned)(bn0 + j) * (unsigned)Kdim + (unsigned)(k0 + kk0);
      async_copy_b128(src,     &Bs[buf][j * LDT + kk0]);
      async_copy_b128(src + 8, &Bs[buf][j * LDT + kk0 + 8]);
    }
  };

  v8f acc[2][2] = {};
  const int nsteps = Kdim >> 5;

  stageA(0, 0);
  stageB(0, 0);
  tile_fence();

  for (int s = 0; s < nsteps; ++s) {
    const int cur = s & 1;
    if (s + 1 < nsteps) {                      // prefetch next tile -> other buffer
      stageA(cur ^ 1, (s + 1) << 5);
      stageB(cur ^ 1, (s + 1) << 5);
    }

    // fragments per ISA 16-bit A(16x32)/B(32x16) layouts
    Frag16 af[2], bfr[2];
#pragma unroll
    for (int mi = 0; mi < 2; ++mi) {
      const unsigned int* ap = reinterpret_cast<const unsigned int*>(
          &As[cur][(wm * 32 + mi * 16 + l16) * LDT]);
#pragma unroll
      for (int j = 0; j < 4; ++j) {            // lanes0-15: K0..7 & K16..23; +8 hi half
        af[mi].u[j]     = ap[half * 4 + j];
        af[mi].u[4 + j] = ap[8 + half * 4 + j];
      }
    }
#pragma unroll
    for (int ni = 0; ni < 2; ++ni) {
      const unsigned int* bp = reinterpret_cast<const unsigned int*>(
          &Bs[cur][(wn * 32 + ni * 16 + l16) * LDT]);
#pragma unroll
      for (int j = 0; j < 4; ++j) {            // lanes0-15: K0..7 & K16..23; +8 hi half
        bfr[ni].u[j]     = bp[half * 4 + j];
        bfr[ni].u[4 + j] = bp[8 + half * 4 + j];
      }
    }
#pragma unroll
    for (int mi = 0; mi < 2; ++mi)
#pragma unroll
      for (int ni = 0; ni < 2; ++ni)
        acc[mi][ni] = __builtin_amdgcn_wmma_f32_16x16x32_bf16(
            false, af[mi].v, false, bfr[ni].v, (short)0, acc[mi][ni], false, false);

    tile_fence();                              // own async done + all waves synced
  }

  // ---- epilogue: C layout VGPR j -> row half*8+j, col l16 ------------------
  const unsigned cbase = (unsigned)b * strideC;
  const float gval = (EPI == 3) ? gammaPtr[0] : 0.0f;
#pragma unroll
  for (int mi = 0; mi < 2; ++mi) {
#pragma unroll
    for (int ni = 0; ni < 2; ++ni) {
      const int col = bn0 + wn * 32 + ni * 16 + l16;
#pragma unroll
      for (int j = 0; j < 8; ++j) {
        const int rloc = wm * 32 + mi * 16 + half * 8 + j;
        const int row  = bm0 + rloc;
        const unsigned idx = cbase + (unsigned)row * (unsigned)Ndim + (unsigned)col;
        float v = acc[mi][ni][j];
        if (EPI == 0) {
          Cb[idx] = f2bf(v * sSc[rloc] + sBi[rloc]);
        } else if (EPI == 1) {
          Cf[idx] = v * constScale;
        } else {
          Cf[idx] = residual[(unsigned)b * strideR + (unsigned)row * (unsigned)Ndim +
                             (unsigned)col] + gval * v;
        }
      }
    }
  }
}

// ---- row softmax over M=1024, fp32 in -> bf16 out -------------------------
__global__ __launch_bounds__(256) void softmax_rows_1024(
    const float* __restrict__ S, unsigned short* __restrict__ St) {
  const unsigned row = blockIdx.x;
  const float* src = S + (size_t)row * 1024;
  unsigned short* dst = St + (size_t)row * 1024;
  const int tid = threadIdx.x;
  __shared__ float red[256];

  float vals[4];
  float mx = -INFINITY;
#pragma unroll
  for (int e = 0; e < 4; ++e) {
    vals[e] = src[tid + e * 256];
    mx = fmaxf(mx, vals[e]);
  }
  red[tid] = mx; __syncthreads();
  for (int s = 128; s > 0; s >>= 1) {
    if (tid < s) red[tid] = fmaxf(red[tid], red[tid + s]);
    __syncthreads();
  }
  mx = red[0]; __syncthreads();

  float sum = 0.0f;
#pragma unroll
  for (int e = 0; e < 4; ++e) { vals[e] = __expf(vals[e] - mx); sum += vals[e]; }
  red[tid] = sum; __syncthreads();
  for (int s = 128; s > 0; s >>= 1) {
    if (tid < s) red[tid] += red[tid + s];
    __syncthreads();
  }
  const float inv = 1.0f / red[0];
#pragma unroll
  for (int e = 0; e < 4; ++e) dst[tid + e * 256] = f2bf(vals[e] * inv);
}

// ---------------------------------------------------------------------------
extern "C" void kernel_launch(void* const* d_in, const int* in_sizes, int n_in,
                              void* d_out, int out_size, void* d_ws, size_t ws_size,
                              hipStream_t stream) {
  (void)in_sizes; (void)n_in; (void)out_size; (void)ws_size;
  const long Bn = 8, C = 1024, N = 4096, M = 1024, P = 512;

  const float* x   = (const float*)d_in[0];
  const float* wq  = (const float*)d_in[1];
  const float* bq  = (const float*)d_in[2];
  const float* wk  = (const float*)d_in[3];
  const float* bk  = (const float*)d_in[4];
  const float* wv  = (const float*)d_in[5];
  const float* bv  = (const float*)d_in[6];
  const float* wo  = (const float*)d_in[7];
  const float* qg  = (const float*)d_in[8];
  const float* qb  = (const float*)d_in[9];
  const float* qm  = (const float*)d_in[10];
  const float* qv  = (const float*)d_in[11];
  const float* kg  = (const float*)d_in[12];
  const float* kb  = (const float*)d_in[13];
  const float* km  = (const float*)d_in[14];
  const float* kv  = (const float*)d_in[15];
  const float* gam = (const float*)d_in[16];
  float* out = (float*)d_out;

  // workspace carve-up (256B aligned)
  char* ws = (char*)d_ws;
  size_t off = 0;
  auto carve = [&](size_t bytes) {
    char* p = ws + off;
    off = (off + bytes + 255) & ~(size_t)255;
    return p;
  };
  unsigned short* xb   = (unsigned short*)carve(Bn * C * N * 2);  // 64 MB
  unsigned short* pb   = (unsigned short*)carve(Bn * C * M * 2);  // 16 MB
  unsigned short* wqb  = (unsigned short*)carve(P * C * 2);
  unsigned short* wkb  = (unsigned short*)carve(P * C * 2);
  unsigned short* wvb  = (unsigned short*)carve(P * C * 2);
  unsigned short* wob  = (unsigned short*)carve(C * P * 2);
  float* sq    = (float*)carve(P * 4);
  float* tq    = (float*)carve(P * 4);
  float* sk    = (float*)carve(P * 4);
  float* tk    = (float*)carve(P * 4);
  float* ones  = (float*)carve(P * 4);
  float* zeros = (float*)carve(P * 4);
  unsigned short* Qb = (unsigned short*)carve(Bn * P * N * 2);    // 32 MB
  unsigned short* Kb = (unsigned short*)carve(Bn * P * M * 2);    // 8 MB
  unsigned short* Vb = (unsigned short*)carve(Bn * P * M * 2);    // 8 MB
  float*          Sf = (float*)carve(Bn * N * M * 4);             // 128 MB
  unsigned short* St = (unsigned short*)carve(Bn * N * M * 2);    // 64 MB
  unsigned short* Ob = (unsigned short*)carve(Bn * P * N * 2);    // 32 MB

  // 1) convert activations & weights to bf16; fused maxpool; fold BN
  cvt_f32_bf16_vec4<<<(Bn * C * N / 4 + 255) / 256, 256, 0, stream>>>(x, xb, (int)(Bn * C * N / 4));
  pool2x2_bf16<<<(Bn * C * M + 255) / 256, 256, 0, stream>>>(x, pb, (int)(Bn * C * M));
  cvt_f32_bf16_vec4<<<(P * C / 4 + 255) / 256, 256, 0, stream>>>(wq, wqb, (int)(P * C / 4));
  cvt_f32_bf16_vec4<<<(P * C / 4 + 255) / 256, 256, 0, stream>>>(wk, wkb, (int)(P * C / 4));
  cvt_f32_bf16_vec4<<<(P * C / 4 + 255) / 256, 256, 0, stream>>>(wv, wvb, (int)(P * C / 4));
  cvt_f32_bf16_vec4<<<(C * P / 4 + 255) / 256, 256, 0, stream>>>(wo, wob, (int)(C * P / 4));
  bn_fold<<<2, 256, 0, stream>>>(qg, qb, qm, qv, bq, sq, tq, (int)P);
  bn_fold<<<2, 256, 0, stream>>>(kg, kb, km, kv, bk, sk, tk, (int)P);
  fill_const<<<2, 256, 0, stream>>>(ones, 1.0f, (int)P);
  fill_const<<<2, 256, 0, stream>>>(zeros, 0.0f, (int)P);

  // 2) Q = BN(Wq·X) [512x4096], K = BN(Wk·Xp) [512x1024], V = Wv·Xp + bv
  gemm_bf16_wmma<false, false, 0><<<dim3(32, 8, 8), 256, 0, stream>>>(
      wqb, xb, Qb, nullptr, 512, 4096, 1024, 0, (unsigned)(C * N), (unsigned)(P * N),
      sq, tq, 0.f, nullptr, 0, nullptr);
  gemm_bf16_wmma<false, false, 0><<<dim3(8, 8, 8), 256, 0, stream>>>(
      wkb, pb, Kb, nullptr, 512, 1024, 1024, 0, (unsigned)(C * M), (unsigned)(P * M),
      sk, tk, 0.f, nullptr, 0, nullptr);
  gemm_bf16_wmma<false, false, 0><<<dim3(8, 8, 8), 256, 0, stream>>>(
      wvb, pb, Vb, nullptr, 512, 1024, 1024, 0, (unsigned)(C * M), (unsigned)(P * M),
      ones, bv, 0.f, nullptr, 0, nullptr);

  // 3) S = Q^T·K / sqrt(P)   [4096 x 1024]  (TN)
  gemm_bf16_wmma<true, false, 1><<<dim3(8, 64, 8), 256, 0, stream>>>(
      Qb, Kb, nullptr, Sf, 4096, 1024, 512, (unsigned)(P * N), (unsigned)(P * M),
      (unsigned)(N * M), ones, zeros, 0.0441941738241592f, nullptr, 0, nullptr);

  // 4) softmax rows -> bf16
  softmax_rows_1024<<<Bn * N, 256, 0, stream>>>(Sf, St);

  // 5) O = V·St^T   [512 x 4096]  (NT, async B staging)
  gemm_bf16_wmma<false, true, 0><<<dim3(32, 8, 8), 256, 0, stream>>>(
      Vb, St, Ob, nullptr, 512, 4096, 1024, (unsigned)(P * M), (unsigned)(N * M),
      (unsigned)(P * N), ones, zeros, 0.f, nullptr, 0, nullptr);

  // 6) out = x + gamma * (Wo·O)   [1024 x 4096]
  gemm_bf16_wmma<false, false, 3><<<dim3(32, 16, 8), 256, 0, stream>>>(
      wob, Ob, nullptr, out, 1024, 4096, 512, 0, (unsigned)(P * N), (unsigned)(C * N),
      ones, zeros, 0.f, x, (unsigned)(C * N), gam);
}
